// CausalSelfAttention_27960237097427
// MI455X (gfx1250) — compile-verified, matrix operand order checked by reading, not run on hardware
//
#include <hip/hip_runtime.h>

#define DIM   960
#define NH    15
#define NKV   5
#define HD    64
#define BB    4
#define TT    2048
#define QKV_W 1600            // 960 (q) + 320 (k) + 320 (v)
#define ROWS  (BB * TT)       // 8192

typedef __attribute__((ext_vector_type(16))) __bf16 bf16x16;
typedef __attribute__((ext_vector_type(8)))  float  f32x8;

union frag_cast { uint4 q[2]; bf16x16 v; };

// 16x32 bf16 fragment loader (A-layout per ISA 7.12.2: lane L holds row L%16,
// K = {0..7,16..23} + 8*(L/16)). Also used for B fragments when the tile is
// stored transposed (row = output column, K contiguous). All call sites are
// 16-byte aligned -> two b128 loads.
__device__ __forceinline__ bf16x16 load_frag(const __bf16* base, int stride) {
  int lane = threadIdx.x & 31;
  int row  = lane & 15;
  int half = lane >> 4;
  const __bf16* p = base + row * stride + half * 8;
  frag_cast f;
  f.q[0] = *(const uint4*)(p);
  f.q[1] = *(const uint4*)(p + 16);
  return f.v;
}

__device__ __forceinline__ f32x8 wmma_bf16(bf16x16 a, bf16x16 b, f32x8 c) {
  return __builtin_amdgcn_wmma_f32_16x16x32_bf16(false, a, false, b, (short)0, c,
                                                 false, false);
}

// async global->LDS 16B copy (GLOBAL_LOAD_ASYNC_TO_LDS_B128, ASYNCcnt-tracked)
__device__ __forceinline__ void async_copy16(void* lds_dst, const void* gsrc) {
  unsigned int       la = (unsigned int)(unsigned long long)lds_dst;  // flat low 32 = LDS addr
  unsigned long long ga = (unsigned long long)gsrc;
  asm volatile("global_load_async_to_lds_b128 %0, %1, off" :: "v"(la), "v"(ga) : "memory");
}
__device__ __forceinline__ void async_wait0() {
  asm volatile("s_wait_asynccnt 0" ::: "memory");
}

// ---------------------------------------------------------------- convert ---
__global__ void convert_f32_bf16(const float* __restrict__ src,
                                 __bf16* __restrict__ dst,
                                 int rows, int cols, int dstride) {
  long long i = (long long)blockIdx.x * blockDim.x + threadIdx.x;
  long long n = (long long)rows * cols;
  if (i >= n) return;
  int r = (int)(i / cols);
  int c = (int)(i % cols);
  dst[(long long)r * dstride + c] = (__bf16)src[i];
}

// ------------------------------------------------------------------- rope ---
__global__ void rope_kernel(__bf16* __restrict__ qkv,
                            const float* __restrict__ cosT,
                            const float* __restrict__ sinT) {
  long long i = (long long)blockIdx.x * blockDim.x + threadIdx.x;
  const long long npairs = (long long)ROWS * 640;   // 480 q-pairs + 160 k-pairs
  if (i >= npairs) return;
  int row = (int)(i / 640);
  int pi  = (int)(i % 640);
  int col, fi;
  if (pi < 480) { int h = pi >> 5; fi = pi & 31; col = h * 64 + 2 * fi; }
  else { int pk = pi - 480; int h = pk >> 5; fi = pk & 31; col = 960 + h * 64 + 2 * fi; }
  int t = row & (TT - 1);
  float c = cosT[t * 32 + fi];
  float s = sinT[t * 32 + fi];
  __bf16* p = qkv + (long long)row * QKV_W + col;
  float ur = (float)p[0], ui = (float)p[1];
  p[0] = (__bf16)(ur * c - ui * s);
  p[1] = (__bf16)(ur * s + ui * c);
}

// ------------------------------------------------------------------- gemm ---
// C[M,N] = A[M,K] @ W[K,N]; 128x64 block tile, 128 threads (4 waves), K step
// 32, double-buffered LDS, async A-tile copies. Wave = 32x64 slab (2 row frags).
template <bool STORE_F32>
__global__ __launch_bounds__(128, 1)
void gemm_bf16_wmma(const __bf16* __restrict__ A, const __bf16* __restrict__ W,
                    void* __restrict__ Cv, int K, int lda, int ldw, int ldc) {
  __shared__ __bf16 At[2][128 * 32];   // [m][k], 8KB per buffer
  __shared__ __bf16 Bt[2][64 * 32];    // transposed [n][k], 4KB per buffer
  int tid = threadIdx.x, wave = tid >> 5, lane = tid & 31;
  int mb = blockIdx.x * 128, nb = blockIdx.y * 64;
  f32x8 acc[2][4] = {};

  auto load_tiles = [&](int k0, int buf) {
    // A tile: 128x32 = 4096 bf16, async 16B copies (ASYNCcnt)
#pragma unroll
    for (int i = 0; i < 4; ++i) {
      int idx = tid + i * 128;                 // 0..511
      int row = idx >> 2, kq = (idx & 3) * 8;
      async_copy16(&At[buf][row * 32 + kq],
                   &A[(long long)(mb + row) * lda + k0 + kq]);
    }
    // W tile: 32x64, stored transposed (element scatter -> manual)
#pragma unroll
    for (int i = 0; i < 2; ++i) {
      int idx = tid + i * 128;                 // 0..255
      int k = idx >> 3, n0 = (idx & 7) * 8;
      frag_cast w; w.q[0] = *(const uint4*)&W[(long long)(k0 + k) * ldw + nb + n0];
      const __bf16* wv = (const __bf16*)&w.q[0];
#pragma unroll
      for (int j = 0; j < 8; ++j) Bt[buf][(n0 + j) * 32 + k] = wv[j];
    }
    if (k0 + 32 < K) {                         // prefetch tile after next
      __builtin_prefetch(&A[(long long)(mb + (tid >> 2)) * lda + k0 + 32], 0, 1);
      __builtin_prefetch(&W[(long long)(k0 + 32 + (tid >> 3)) * ldw + nb], 0, 1);
    }
  };

  load_tiles(0, 0);
  async_wait0();
  __syncthreads();
  int niter = K / 32;
  for (int it = 0; it < niter; ++it) {
    int cur = it & 1;
    if (it + 1 < niter) load_tiles((it + 1) * 32, cur ^ 1);
    bf16x16 a0 = load_frag(&At[cur][(wave * 32) * 32], 32);
    bf16x16 a1 = load_frag(&At[cur][(wave * 32 + 16) * 32], 32);
#pragma unroll
    for (int c = 0; c < 4; ++c) {
      bf16x16 b = load_frag(&Bt[cur][c * 16 * 32], 32);
      acc[0][c] = wmma_bf16(a0, b, acc[0][c]);
      acc[1][c] = wmma_bf16(a1, b, acc[1][c]);
    }
    async_wait0();
    __syncthreads();
  }

  int half = lane >> 4, coll = lane & 15;
#pragma unroll
  for (int s = 0; s < 2; ++s) {
    int row0 = mb + wave * 32 + s * 16 + half * 8;
    int col0 = nb + coll;
    if (STORE_F32) {
      float* C = (float*)Cv;
#pragma unroll
      for (int c = 0; c < 4; ++c)
#pragma unroll
        for (int r = 0; r < 8; ++r)
          C[(long long)(row0 + r) * ldc + col0 + c * 16] = acc[s][c][r];
    } else {
      __bf16* C = (__bf16*)Cv;
#pragma unroll
      for (int c = 0; c < 4; ++c)
#pragma unroll
        for (int r = 0; r < 8; ++r)
          C[(long long)(row0 + r) * ldc + col0 + c * 16] = (__bf16)acc[s][c][r];
    }
  }
}

// -------------------------------------------------------- flash attention ---
// Block: 256 threads = 8 waves; each wave owns 16 query rows (block = 128 rows)
// of one (batch b, q-head h). 32-key tiles, double-buffered K/V, online softmax.
// Row sums via an extra WMMA against an all-ones B fragment. Wave-uniform
// gates are pinned to SGPRs (readfirstlane) so they lower to scalar branches.
__global__ __launch_bounds__(256, 1)
void attn_kernel(const __bf16* __restrict__ qkv, __bf16* __restrict__ aout) {
  __shared__ __bf16 Kt[2][32 * 64];    // [key][d]   (acts as Bt for Q@K^T)
  __shared__ __bf16 Vt[2][64 * 32];    // [d][key]   (acts as Bt for P@V)
  __shared__ __bf16 Pl[8][16 * 32];    // per-wave P tile [row][key]

  int tid = threadIdx.x, wave = tid >> 5, lane = tid & 31;
  int half = lane >> 4, coll = lane & 15;
  int h  = blockIdx.y;            // q head 0..14
  int b  = blockIdx.z;
  int kh = h / 3;                 // kv head (G = 3)
  int qb = blockIdx.x * 128;
  long long rowbase = (long long)b * TT;
  int wqbase = qb + wave * 16;
  int wq_s = __builtin_amdgcn_readfirstlane(wqbase);  // SGPR copy for gates
  int ntiles = qb / 32 + 4;

  auto load_kv = [&](int sbase, int buf) {
    int key = tid >> 3, d0 = (tid & 7) * 8;    // 256 thr x 8 elems = full tile
    long long grow = (rowbase + sbase + key) * QKV_W + kh * 64 + d0;
    async_copy16(&Kt[buf][key * 64 + d0], &qkv[grow + 960]);   // K: contiguous
    frag_cast vv; vv.q[0] = *(const uint4*)&qkv[grow + 1280];  // V: transpose
    const __bf16* vp = (const __bf16*)&vv.q[0];
#pragma unroll
    for (int j = 0; j < 8; ++j) Vt[buf][(d0 + j) * 32 + key] = vp[j];
    if (sbase + 32 < ntiles * 32) {            // prefetch tile after next
      __builtin_prefetch(&qkv[grow + 960 + 32 * QKV_W], 0, 1);
      __builtin_prefetch(&qkv[grow + 1280 + 32 * QKV_W], 0, 1);
    }
  };

  // Q fragments (16 rows x 64 dims), kept in registers for the whole loop.
  const __bf16* qbase = qkv + (rowbase + wqbase) * QKV_W + h * 64;
  bf16x16 aq0 = load_frag(qbase, QKV_W);
  bf16x16 aq1 = load_frag(qbase + 32, QKV_W);
  bf16x16 ones;
#pragma unroll
  for (int j = 0; j < 16; ++j) ones[j] = (__bf16)1.0f;

  float m[8], l[8];
#pragma unroll
  for (int r = 0; r < 8; ++r) { m[r] = -3.0e38f; l[r] = 0.0f; }
  f32x8 acc[4] = {};

  load_kv(0, 0);
  async_wait0();
  __syncthreads();
  for (int kt = 0; kt < ntiles; ++kt) {
    int cur = kt & 1;
    int sbase = kt * 32;
    if (kt + 1 < ntiles) load_kv(sbase + 32, cur ^ 1);

    if (sbase <= wq_s + 15) {                  // scalar-branch causal gate
      f32x8 sc[2];
#pragma unroll
      for (int nh = 0; nh < 2; ++nh) {
        f32x8 z = {};
        z = wmma_bf16(aq0, load_frag(&Kt[cur][nh * 16 * 64], 64), z);
        z = wmma_bf16(aq1, load_frag(&Kt[cur][nh * 16 * 64 + 32], 64), z);
        sc[nh] = z;
      }

      float es[8];
      if (sbase + 31 > wq_s) {                 // diagonal tile: apply mask
#pragma unroll
        for (int r = 0; r < 8; ++r) {
          int qrow = wqbase + r + 8 * half;
          float s0 = sc[0][r] * 0.125f;
          float s1 = sc[1][r] * 0.125f;
          if (sbase + coll > qrow)      s0 = -3.0e38f;
          if (sbase + 16 + coll > qrow) s1 = -3.0e38f;
          sc[0][r] = s0; sc[1][r] = s1;
          float mx = fmaxf(s0, s1);
          mx = fmaxf(mx, __shfl_xor(mx, 1, 16));
          mx = fmaxf(mx, __shfl_xor(mx, 2, 16));
          mx = fmaxf(mx, __shfl_xor(mx, 4, 16));
          mx = fmaxf(mx, __shfl_xor(mx, 8, 16));
          float mn = fmaxf(m[r], mx);
          es[r] = __expf(m[r] - mn);
          m[r] = mn;
        }
      } else {                                 // interior tile: no mask
#pragma unroll
        for (int r = 0; r < 8; ++r) {
          float s0 = sc[0][r] * 0.125f;
          float s1 = sc[1][r] * 0.125f;
          sc[0][r] = s0; sc[1][r] = s1;
          float mx = fmaxf(s0, s1);
          mx = fmaxf(mx, __shfl_xor(mx, 1, 16));
          mx = fmaxf(mx, __shfl_xor(mx, 2, 16));
          mx = fmaxf(mx, __shfl_xor(mx, 4, 16));
          mx = fmaxf(mx, __shfl_xor(mx, 8, 16));
          float mn = fmaxf(m[r], mx);
          es[r] = __expf(m[r] - mn);
          m[r] = mn;
        }
      }
#pragma unroll
      for (int r = 0; r < 8; ++r) {
        float p0 = __expf(sc[0][r] - m[r]);
        float p1 = __expf(sc[1][r] - m[r]);
        Pl[wave][(r + 8 * half) * 32 + coll]      = (__bf16)p0;
        Pl[wave][(r + 8 * half) * 32 + 16 + coll] = (__bf16)p1;
#pragma unroll
        for (int c = 0; c < 4; ++c) acc[c][r] *= es[r];
      }
      asm volatile("s_wait_dscnt 0" ::: "memory");   // wave-local LDS RAW on Pl

      bf16x16 ap = load_frag(Pl[wave], 32);
      f32x8 z = {};
      f32x8 rs = wmma_bf16(ap, ones, z);             // row sums via WMMA
#pragma unroll
      for (int r = 0; r < 8; ++r) l[r] = l[r] * es[r] + rs[r];
#pragma unroll
      for (int c = 0; c < 4; ++c) {
        bf16x16 bv = load_frag(&Vt[cur][c * 16 * 32], 32);
        acc[c] = wmma_bf16(ap, bv, acc[c]);
      }
    }
    async_wait0();
    __syncthreads();   // next-tile stores complete; current buffers free
  }

  // epilogue: normalize and store bf16 rows into attn buffer [8192][960]
#pragma unroll
  for (int r = 0; r < 8; ++r) {
    float inv = 1.0f / l[r];
    long long row = rowbase + wqbase + r + 8 * half;
#pragma unroll
    for (int c = 0; c < 4; ++c)
      aout[row * DIM + h * 64 + c * 16 + coll] = (__bf16)(acc[c][r] * inv);
  }
}

// ------------------------------------------------------------------ launch ---
extern "C" void kernel_launch(void* const* d_in, const int* in_sizes, int n_in,
                              void* d_out, int out_size, void* d_ws, size_t ws_size,
                              hipStream_t stream) {
  (void)in_sizes; (void)n_in; (void)out_size;
  const float* x  = (const float*)d_in[0];
  const float* fc = (const float*)d_in[1];
  const float* fs = (const float*)d_in[2];
  const float* wq = (const float*)d_in[3];
  const float* wk = (const float*)d_in[4];
  const float* wv = (const float*)d_in[5];
  const float* wo = (const float*)d_in[6];

  char* ws = (char*)d_ws;
  size_t off = 0;
  auto carve = [&](size_t bytes) -> char* {
    char* p = ws + off;
    off += (bytes + 255) & ~(size_t)255;
    return p;
  };
  __bf16* xb   = (__bf16*)carve((size_t)ROWS * DIM * 2);     // x in bf16
  __bf16* wqkv = (__bf16*)carve((size_t)DIM * QKV_W * 2);    // packed wq|wk|wv
  __bf16* wob  = (__bf16*)carve((size_t)DIM * DIM * 2);      // wo in bf16
  __bf16* qkv  = (__bf16*)carve((size_t)ROWS * QKV_W * 2);   // q|k|v activations
  __bf16* attn = (__bf16*)carve((size_t)ROWS * DIM * 2);     // attention output
  if (off > ws_size) return;  // workspace too small; nothing safe to do

  auto conv = [&](const float* s, __bf16* d, int rows, int cols, int dstride) {
    long long n = (long long)rows * cols;
    int blocks = (int)((n + 255) / 256);
    convert_f32_bf16<<<blocks, 256, 0, stream>>>(s, d, rows, cols, dstride);
  };
  conv(x,  xb,          ROWS, DIM, DIM);
  conv(wq, wqkv,        DIM,  DIM, QKV_W);
  conv(wk, wqkv + 960,  DIM,  320, QKV_W);
  conv(wv, wqkv + 1280, DIM,  320, QKV_W);
  conv(wo, wob,         DIM,  DIM, DIM);

  // qkv = xb @ [wq|wk|wv]   (M=8192, N=1600, K=960)
  gemm_bf16_wmma<false><<<dim3(ROWS / 128, QKV_W / 64), 128, 0, stream>>>(
      xb, wqkv, qkv, DIM, DIM, QKV_W, QKV_W);

  {
    long long np = (long long)ROWS * 640;
    rope_kernel<<<(int)((np + 255) / 256), 256, 0, stream>>>(qkv, fc, fs);
  }

  attn_kernel<<<dim3(TT / 128, NH, BB), 256, 0, stream>>>(qkv, attn);

  // d_out = attn @ wo   (M=8192, N=960, K=960), fp32 store
  gemm_bf16_wmma<true><<<dim3(ROWS / 128, DIM / 64), 128, 0, stream>>>(
      attn, wob, d_out, DIM, DIM, DIM, DIM);
}
